// Neuralsymbol_7155415515794
// MI455X (gfx1250) — matrix operand order repaired and checked by reading, not compiled, not run on hardware
//
#include <hip/hip_runtime.h>

typedef __attribute__((ext_vector_type(16))) __bf16 v16bf;
typedef __attribute__((ext_vector_type(8)))  __bf16 v8bf;
typedef __attribute__((ext_vector_type(8)))  float  v8f;

constexpr int Bq    = 64;    // batch
constexpr int Tq    = 256;   // time steps
constexpr int Vq    = 512;   // vocab dim
constexpr int NSq   = 64;    // N_SYM
constexpr int NSTKq = 8;     // stack depth
constexpr int BT    = 16;    // batch tile per block (one WMMA M tile)
constexpr int NOUTq = 515;   // 3 + V
constexpr int BDIM  = 256;   // 8 waves

// Dynamic LDS layout size (bytes)
constexpr size_t SMEM_BYTES =
    (size_t)BT * Vq * 2 * 3      // sCtrl, sWork, sInp (bf16, unnormalized)
  + (size_t)BT * 192 * 2         // sG  = gelu([c|w|i]) bf16
  + (size_t)BT * Vq * 4 * 2      // sNewC, sNewW (f32 tanh outputs)
  + (size_t)BT * NSq * 2         // sHf = sigmoid(hf) bf16
  + (size_t)BT * 8 * 4 * 4       // sPtrC, sPtrW, sNPtrC, sNPtrW
  + (size_t)BT * 4 * 4 * 2       // sGateC, sGateW ([16][4] f32)
  + 48 * 4                       // sRinv [3][16]
  + 3 * 256 * 4;                 // sPart [3][16][16]

static __device__ __forceinline__ float sigm(float x) { return 1.f / (1.f + expf(-x)); }

// A fragment (16x32 bf16, M x K) from row-major [M][ldk] bf16.
// Lane L: row = L&15, two 8-elem chunks at k = 8*(L>>4) and +16.
static __device__ __forceinline__ v16bf load_frag_a(const __bf16* base, int ldk, int lane) {
  int row = lane & 15, h = lane >> 4;
  const __bf16* p = base + row * ldk + 8 * h;
  v16bf out;
  *((v8bf*)&out)       = *((const v8bf*)p);
  *(((v8bf*)&out) + 1) = *((const v8bf*)(p + 16));
  return out;
}

// B fragment (32x16 bf16, K x N) from transposed row-major Bt[N][ldk] bf16.
// Lane L: col = L&15, two 8-elem chunks at k = 8*(L>>4) and +16.
static __device__ __forceinline__ v16bf load_frag_b(const __bf16* baseT, int ldk, int lane) {
  int col = lane & 15, h = lane >> 4;
  const __bf16* p = baseT + col * ldk + 8 * h;
  v16bf out;
  *((v8bf*)&out)       = *((const v8bf*)p);
  *(((v8bf*)&out) + 1) = *((const v8bf*)(p + 16));
  return out;
}

static __device__ __forceinline__ v8f wmma_bf16(v16bf a, v16bf b, v8f c) {
  return __builtin_amdgcn_wmma_f32_16x16x32_bf16(false, a, false, b, (short)0, c, false, false);
}

// ---------------- Preprocessing ----------------

// Compute w_hat = sym_w / max(||col||, EPS); emit whT [64][512] (Bt for sims)
// and wh [512][64] (Bt for out projection), both bf16.
__global__ void prep_sym_kernel(const float* __restrict__ sym_w,
                                __bf16* __restrict__ whT, __bf16* __restrict__ wh) {
  __shared__ float rwn[NSq];
  int tid = threadIdx.x;
  if (tid < NSq) {
    float ss = 0.f;
    for (int v = 0; v < Vq; ++v) { float t = sym_w[v * NSq + tid]; ss += t * t; }
    rwn[tid] = 1.f / fmaxf(sqrtf(ss), 1e-8f);
  }
  __syncthreads();
  for (int i = tid; i < Vq * NSq; i += BDIM) {
    int v = i >> 6, n = i & 63;
    float val = sym_w[i] * rwn[n];
    wh[i]            = (__bf16)val;     // wh[v][n]
    whT[n * Vq + v]  = (__bf16)val;     // whT[n][v]
  }
}

// Transpose + pad + bf16-convert the three weight matrices.
// wcT [528][128], wwT [528][192], wfT [64][128].
__global__ void prep_w_kernel(const float* __restrict__ wc, const float* __restrict__ ww,
                              const float* __restrict__ wf,
                              __bf16* __restrict__ wcT, __bf16* __restrict__ wwT,
                              __bf16* __restrict__ wfT) {
  int i = blockIdx.x * blockDim.x + threadIdx.x;
  if (i < 528 * 128) { int n = i >> 7, k = i & 127;
    wcT[i] = (__bf16)((n < NOUTq) ? wc[k * NOUTq + n] : 0.f); }
  if (i < 528 * 192) { int n = i / 192, k = i - n * 192;
    wwT[i] = (__bf16)((n < NOUTq) ? ww[k * NOUTq + n] : 0.f); }
  if (i < 64 * 128)  { int n = i >> 7, k = i & 127;
    wfT[i] = (__bf16)(wf[k * 64 + n]); }
}

// ---------------- Main sequential scan (4 independent blocks x 16 batch rows) ----------------

__global__ __launch_bounds__(BDIM)
void scan_kernel(const float* __restrict__ x,
                 const float* __restrict__ b_control,
                 const float* __restrict__ b_work,
                 const float* __restrict__ b_ff,
                 const float* __restrict__ control_sharp,
                 const float* __restrict__ work_sharp,
                 const __bf16* __restrict__ whT,
                 const __bf16* __restrict__ wh,
                 const __bf16* __restrict__ wcT,
                 const __bf16* __restrict__ wwT,
                 const __bf16* __restrict__ wfT,
                 float* __restrict__ cstk,
                 float* __restrict__ wstk,
                 float* __restrict__ out) {
  extern __shared__ char smem[];
  __bf16* sCtrl  = (__bf16*)smem;                 // [16][512]
  __bf16* sWork  = sCtrl + BT * Vq;
  __bf16* sInp   = sWork + BT * Vq;
  __bf16* sG     = sInp + BT * Vq;                // [16][192]
  float*  sNewC  = (float*)(sG + BT * 192);       // [16][512]
  float*  sNewW  = sNewC + BT * Vq;
  __bf16* sHf    = (__bf16*)(sNewW + BT * Vq);    // [16][64]
  float*  sPtrC  = (float*)(sHf + BT * NSq);      // [16][8]
  float*  sPtrW  = sPtrC + BT * 8;
  float*  sNPtrC = sPtrW + BT * 8;
  float*  sNPtrW = sNPtrC + BT * 8;
  float*  sGateC = sNPtrW + BT * 8;               // [16][4]
  float*  sGateW = sGateC + BT * 4;
  float*  sRinv  = sGateW + BT * 4;               // [3][16]
  float*  sPart  = sRinv + 48;                    // [3][16][16]

  const int tid   = threadIdx.x;
  const int lane  = tid & 31;
  const int wave  = tid >> 5;
  const int bbase = blockIdx.x * BT;

  // ---- init stacks (ZERO_OFFSET) and pointers ----
  {
    float4 fill; fill.x = fill.y = fill.z = fill.w = 1e-5f;
    float4* c4 = (float4*)(cstk + (size_t)bbase * NSTKq * Vq);
    float4* w4 = (float4*)(wstk + (size_t)bbase * NSTKq * Vq);
    const int tot4 = BT * NSTKq * Vq / 4;
    for (int i = tid; i < tot4; i += BDIM) { c4[i] = fill; w4[i] = fill; }
    if (tid < BT * 8) { sPtrC[tid] = ((tid & 7) == 0) ? 1.f : 0.f;
                        sPtrW[tid] = ((tid & 7) == 0) ? 1.f : 0.f; }
  }
  __syncthreads();

  const float csharp = control_sharp[0];
  const float wsharp = work_sharp[0];

  for (int t = 0; t < Tq; ++t) {
    // ---------- Phase 1: soft pop (weighted sum) + input load + sumsq ----------
    {
      int row = tid >> 4, sub = tid & 15;       // 16 rows x 16 lanes, 32 cols each
      int b = bbase + row;
      const float* xrow = x + ((size_t)b * Tq + t) * Vq;
      const float* crow = cstk + (size_t)b * NSTKq * Vq;
      const float* wrow = wstk + (size_t)b * NSTKq * Vq;
      float p0c[8], p0w[8];
      for (int s = 0; s < 8; ++s) { p0c[s] = sPtrC[row * 8 + s]; p0w[s] = sPtrW[row * 8 + s]; }
      float ssc = 0.f, ssw = 0.f, ssi = 0.f;
      for (int c4i = 0; c4i < 8; ++c4i) {
        int col = sub * 32 + c4i * 4;
        float4 ac; ac.x = ac.y = ac.z = ac.w = 0.f;
        float4 aw = ac;
        for (int s = 0; s < 8; ++s) {
          float4 vc = *(const float4*)(crow + s * Vq + col);
          float4 vw = *(const float4*)(wrow + s * Vq + col);
          ac.x += p0c[s] * vc.x; ac.y += p0c[s] * vc.y; ac.z += p0c[s] * vc.z; ac.w += p0c[s] * vc.w;
          aw.x += p0w[s] * vw.x; aw.y += p0w[s] * vw.y; aw.z += p0w[s] * vw.z; aw.w += p0w[s] * vw.w;
        }
        float4 xi = *(const float4*)(xrow + col);
        ssc += ac.x*ac.x + ac.y*ac.y + ac.z*ac.z + ac.w*ac.w;
        ssw += aw.x*aw.x + aw.y*aw.y + aw.z*aw.z + aw.w*aw.w;
        ssi += xi.x*xi.x + xi.y*xi.y + xi.z*xi.z + xi.w*xi.w;
        sCtrl[row*Vq+col+0]=(__bf16)ac.x; sCtrl[row*Vq+col+1]=(__bf16)ac.y;
        sCtrl[row*Vq+col+2]=(__bf16)ac.z; sCtrl[row*Vq+col+3]=(__bf16)ac.w;
        sWork[row*Vq+col+0]=(__bf16)aw.x; sWork[row*Vq+col+1]=(__bf16)aw.y;
        sWork[row*Vq+col+2]=(__bf16)aw.z; sWork[row*Vq+col+3]=(__bf16)aw.w;
        sInp [row*Vq+col+0]=(__bf16)xi.x; sInp [row*Vq+col+1]=(__bf16)xi.y;
        sInp [row*Vq+col+2]=(__bf16)xi.z; sInp [row*Vq+col+3]=(__bf16)xi.w;
      }
      sPart[0*256 + row*16 + sub] = ssc;
      sPart[1*256 + row*16 + sub] = ssw;
      sPart[2*256 + row*16 + sub] = ssi;
    }
    __syncthreads();
    if (tid < 48) {
      int src = tid >> 4, row = tid & 15;
      float s = 0.f;
      for (int k = 0; k < 16; ++k) s += sPart[src*256 + row*16 + k];
      sRinv[src*16 + row] = 1.f / fmaxf(sqrtf(s), 1e-8f);
    }
    __syncthreads();

    // ---------- Phase 2: sims = norm(v) @ w_hat  (3 sources x 4 N-tiles), gelu -> sG ----------
    for (int id = wave; id < 12; id += 8) {
      int src = id >> 2, nt = id & 3;
      const __bf16* A = (src == 0) ? sCtrl : (src == 1) ? sWork : sInp;
      v8f acc = {0.f,0.f,0.f,0.f,0.f,0.f,0.f,0.f};
      for (int kt = 0; kt < 16; ++kt) {
        v16bf a = load_frag_a(A + kt * 32, Vq, lane);
        v16bf b = load_frag_b(whT + (size_t)(nt * 16) * Vq + kt * 32, Vq, lane);
        acc = wmma_bf16(a, b, acc);
      }
      int h = lane >> 4, cn = lane & 15;
      int gcol = src * 64 + nt * 16 + cn;
      for (int r = 0; r < 8; ++r) {
        int m = r + 8 * h;
        float z = acc[r] * sRinv[src * 16 + m];
        float ge = 0.5f * z * (1.f + erff(z * 0.70710678118654752f));
        sG[m * 192 + gcol] = (__bf16)ge;
      }
    }
    __syncthreads();

    // ---------- Phase 3: hc (33 tiles, K=128 gathered), hw (33, K=192), hf (4, K=128) ----------
    for (int id = wave; id < 70; id += 8) {
      if (id < 33) {            // hc: A = g[c|i], B = wcT
        v8f acc = {0.f,0.f,0.f,0.f,0.f,0.f,0.f,0.f};
        for (int kt = 0; kt < 4; ++kt) {
          int koff = (kt < 2) ? kt * 32 : kt * 32 + 64;   // skip w block
          v16bf a = load_frag_a(sG + koff, 192, lane);
          v16bf b = load_frag_b(wcT + (size_t)(id * 16) * 128 + kt * 32, 128, lane);
          acc = wmma_bf16(a, b, acc);
        }
        int h = lane >> 4, cn = lane & 15, n = id * 16 + cn;
        if (n < NOUTq) {
          float bias = b_control[n];
          for (int r = 0; r < 8; ++r) {
            int m = r + 8 * h;
            float v = acc[r] + bias;
            if (n < 3) sGateC[m * 4 + n] = sigm(v);
            else       sNewC[m * Vq + (n - 3)] = tanhf(v);
          }
        }
      } else if (id < 66) {     // hw: A = g full, B = wwT
        int nt = id - 33;
        v8f acc = {0.f,0.f,0.f,0.f,0.f,0.f,0.f,0.f};
        for (int kt = 0; kt < 6; ++kt) {
          v16bf a = load_frag_a(sG + kt * 32, 192, lane);
          v16bf b = load_frag_b(wwT + (size_t)(nt * 16) * 192 + kt * 32, 192, lane);
          acc = wmma_bf16(a, b, acc);
        }
        int h = lane >> 4, cn = lane & 15, n = nt * 16 + cn;
        if (n < NOUTq) {
          float bias = b_work[n];
          for (int r = 0; r < 8; ++r) {
            int m = r + 8 * h;
            float v = acc[r] + bias;
            if (n < 3) sGateW[m * 4 + n] = sigm(v);
            else       sNewW[m * Vq + (n - 3)] = tanhf(v);
          }
        }
      } else {                  // hf: A = g[c|w], B = wfT; sigmoid -> bf16
        int nt = id - 66;
        v8f acc = {0.f,0.f,0.f,0.f,0.f,0.f,0.f,0.f};
        for (int kt = 0; kt < 4; ++kt) {
          v16bf a = load_frag_a(sG + kt * 32, 192, lane);
          v16bf b = load_frag_b(wfT + (size_t)(nt * 16) * 128 + kt * 32, 128, lane);
          acc = wmma_bf16(a, b, acc);
        }
        int h = lane >> 4, cn = lane & 15, n = nt * 16 + cn;
        float bias = b_ff[n];
        for (int r = 0; r < 8; ++r) {
          int m = r + 8 * h;
          sHf[m * NSq + n] = (__bf16)sigm(acc[r] + bias);
        }
      }
    }
    __syncthreads();

    // ---------- Phase 4a: out = sigmoid(hf) @ w_hat^T -> global ----------
    for (int id = wave; id < 32; id += 8) {
      v8f acc = {0.f,0.f,0.f,0.f,0.f,0.f,0.f,0.f};
      for (int kt = 0; kt < 2; ++kt) {
        v16bf a = load_frag_a(sHf + kt * 32, NSq, lane);
        v16bf b = load_frag_b(wh + (size_t)(id * 16) * NSq + kt * 32, NSq, lane);
        acc = wmma_bf16(a, b, acc);
      }
      int h = lane >> 4, cn = lane & 15, n = id * 16 + cn;
      for (int r = 0; r < 8; ++r) {
        int m = r + 8 * h;
        out[((size_t)(bbase + m) * Tq + t) * Vq + n] = acc[r];
      }
    }
    // ---------- Phase 4b: sharpened (unnormalized) new pointers ----------
    {
      int st = tid >> 7, r8 = tid & 127, row = r8 >> 3, s = r8 & 7;
      float* P = st ? sPtrW : sPtrC;
      float* G = st ? sGateW : sGateC;
      float sharp = st ? wsharp : csharp;
      float a = G[row*4+0], bg = G[row*4+1], cg = G[row*4+2];
      float u = a * P[row*8 + s] + bg * P[row*8 + ((s+2)&7)] + cg * P[row*8 + ((s+1)&7)];
      float p = powf(fmaxf(u, 0.f) + 1e-10f, sharp);
      (st ? sNPtrW : sNPtrC)[row*8 + s] = p;
    }
    __syncthreads();

    // ---------- Phase 5: fused stack RMW: stk = stk*(1-p0)*blend + a*p0*val ----------
    {
      const int total = 2 * BT * NSTKq * (Vq / 4);   // 32768 float4 ops
      for (int i = tid; i < total; i += BDIM) {
        int st  = i >> 14;
        int rem = i & 16383;
        int row = rem >> 10;
        int rm2 = rem & 1023;
        int s   = rm2 >> 7;
        int c4i = rm2 & 127;
        float* P   = st ? sPtrW : sPtrC;
        float* G   = st ? sGateW : sGateC;
        float* VAL = st ? sNewW : sNewC;
        float* STK = st ? wstk : cstk;
        float a  = G[row*4+0], bg = G[row*4+1], cg = G[row*4+2];
        float p0 = P[row*8 + s];
        float p1 = P[row*8 + ((s+1)&7)];
        float f  = (1.f - p0) * (a * (1.f - p0) + bg * (1.f - p1) + cg);
        float av = a * p0;
        size_t base = ((size_t)(bbase + row) * NSTKq + s) * Vq + c4i * 4;
        float4 old = *(float4*)(STK + base);
        float4 vl  = *(const float4*)(VAL + row * Vq + c4i * 4);
        float4 nw;
        nw.x = old.x * f + av * vl.x; nw.y = old.y * f + av * vl.y;
        nw.z = old.z * f + av * vl.z; nw.w = old.w * f + av * vl.w;
        *(float4*)(STK + base) = nw;
      }
    }
    __syncthreads();

    // ---------- Phase 6: normalize pointers in place ----------
    {
      int st = tid >> 7, r8 = tid & 127, row = r8 >> 3, s = r8 & 7;
      float* NP = st ? sNPtrW : sNPtrC;
      float* P  = st ? sPtrW  : sPtrC;
      float sum = 0.f;
      for (int k = 0; k < 8; ++k) sum += NP[row*8 + k];
      P[row*8 + s] = NP[row*8 + s] / fmaxf(sum, 1e-8f);
    }
    __syncthreads();
  }
}

// ---------------- Host launcher ----------------

extern "C" void kernel_launch(void* const* d_in, const int* in_sizes, int n_in,
                              void* d_out, int out_size, void* d_ws, size_t ws_size,
                              hipStream_t stream) {
  const float* x         = (const float*)d_in[0];
  const float* sym_w     = (const float*)d_in[1];
  const float* w_control = (const float*)d_in[2];
  const float* b_control = (const float*)d_in[3];
  const float* w_work    = (const float*)d_in[4];
  const float* b_work    = (const float*)d_in[5];
  const float* w_ff      = (const float*)d_in[6];
  const float* b_ff      = (const float*)d_in[7];
  const float* c_sharp   = (const float*)d_in[8];
  const float* w_sharp   = (const float*)d_in[9];
  float* out = (float*)d_out;

  char* ws = (char*)d_ws;
  size_t off = 0;
  auto alloc = [&](size_t bytes) -> void* {
    void* p = ws + off;
    off = (off + bytes + 255) & ~(size_t)255;
    return p;
  };
  float*  cstk = (float*)alloc((size_t)Bq * NSTKq * Vq * sizeof(float));   // 1 MB
  float*  wstk = (float*)alloc((size_t)Bq * NSTKq * Vq * sizeof(float));   // 1 MB
  __bf16* whT  = (__bf16*)alloc((size_t)NSq * Vq * 2);
  __bf16* wh   = (__bf16*)alloc((size_t)Vq * NSq * 2);
  __bf16* wcT  = (__bf16*)alloc((size_t)528 * 128 * 2);
  __bf16* wwT  = (__bf16*)alloc((size_t)528 * 192 * 2);
  __bf16* wfT  = (__bf16*)alloc((size_t)64 * 128 * 2);

  prep_sym_kernel<<<1, BDIM, 0, stream>>>(sym_w, whT, wh);
  int prep_n = 528 * 192;
  prep_w_kernel<<<(prep_n + BDIM - 1) / BDIM, BDIM, 0, stream>>>(
      w_control, w_work, w_ff, wcT, wwT, wfT);

  scan_kernel<<<Bq / BT, BDIM, SMEM_BYTES, stream>>>(
      x, b_control, b_work, b_ff, c_sharp, w_sharp,
      whT, wh, wcT, wwT, wfT, cstk, wstk, out);
}
